// AutoCorrelation_31275951849557
// MI455X (gfx1250) — matrix operand order repaired and checked by reading, not compiled, not run on hardware
//
#include <hip/hip_runtime.h>

// ---- sizes from reference ----
#define Ln   2048
#define Bn   32
#define Hn   16
#define En   64
#define EH   32          // E-split: 32 e-rows per workgroup (2 WGs per (b,h))
#define TOPK 7
#define NPART 4          // partials per (b,h): 2 e-halves * 2 t-halves

// ---- plain clang ext_vector types (union/trivially-copyable safe) ----
typedef __attribute__((ext_vector_type(16))) _Float16     f16x16;
typedef __attribute__((ext_vector_type(8)))  float        f32x8;
typedef __attribute__((ext_vector_type(4)))  float        f32x4;
typedef __attribute__((ext_vector_type(4)))  unsigned int u32x4;
typedef __attribute__((ext_vector_type(2)))  unsigned int u32x2;

union V16 { f16x16 v; u32x4 u[2]; };
union P4  { _Float16 h[4]; u32x2 u; };

// =====================================================================
// Kernel 1: mean-over-E circular cross-correlation via WMMA.
//   partial[wg*2+tHalf][tau] = sum_{t in half, e in slab} q[e,t+tau]*k[e,t]
//   tau = m + 16*n + 256*T.  Each wave owns two tiles T={2p,2p+1}: the A
//   operand (q rows) is shared -> 2 A-loads + 4 B-loads per 2 WMMAs.
//   Loop is uniform (constant bounds, scalar control); wave-dependent
//   offsets are folded into per-lane base rows.  unroll 4 rotates enough
//   operand sets to cover the 4-coexec WMMA->VALU WAR hazard without nops.
//   A[m, kk=e] = q_lds[(t0+m)&2047][e]              (16B-aligned b128)
//   B[kk=e, n] = k_lds[(t0-16n-256T)&2047][e]       (16B-aligned b128)
// =====================================================================
__launch_bounds__(256, 1)
__global__ void corr_wmma_kernel(const float* __restrict__ Q,
                                 const float* __restrict__ Kk,
                                 float* __restrict__ meanPart) {
  __shared__ _Float16 qs[Ln * EH];   // 128 KB
  __shared__ _Float16 ks[Ln * EH];   // 128 KB  (WGP allows 320 KB)

  const int wg = blockIdx.x;         // 0 .. B*H*2-1
  const int eh = wg & 1;
  const int bh = wg >> 1;
  const int h  = bh & (Hn - 1);
  const int b  = bh >> 4;

  const int tid = threadIdx.x;

  // ---- stage q,k slab into LDS as f16, layout [t][e] (e contiguous) ----
  {
    const int e4 = (tid & 7) * 4;             // 0..28
    const int tr = tid >> 3;                  // 0..31
    const size_t base = ((size_t)b * Ln) * (Hn * En)
                      + (size_t)h * En + (size_t)(eh * EH) + e4;
    #pragma unroll 2
    for (int t = tr; t < Ln; t += 32) {
      const f32x4 fq = *(const f32x4*)(Q  + base + (size_t)t * (Hn * En));
      const f32x4 fk = *(const f32x4*)(Kk + base + (size_t)t * (Hn * En));
      P4 pq, pk;
      pq.h[0] = (_Float16)fq.x; pq.h[1] = (_Float16)fq.y;
      pq.h[2] = (_Float16)fq.z; pq.h[3] = (_Float16)fq.w;
      pk.h[0] = (_Float16)fk.x; pk.h[1] = (_Float16)fk.y;
      pk.h[2] = (_Float16)fk.z; pk.h[3] = (_Float16)fk.w;
      *(u32x2*)(qs + t * EH + e4) = pq.u;
      *(u32x2*)(ks + t * EH + e4) = pk.u;
    }
  }
  __syncthreads();

  // ---- per-wave WMMA accumulation ----
  const int lane  = tid & 31;
  const int wv    = tid >> 5;          // 8 waves
  const int p     = wv & 3;            // tile pair -> tiles 2p, 2p+1
  const int tH    = wv >> 2;           // t-half 0/1
  const int lm    = lane & 15;         // A row-shift m  /  B column n
  const int gHi   = lane >> 4;         // K-half select (contiguous k order, same for A and B)
  const int eA    = gHi * 16;          // A: lanes<16 K 0..15, else 16..31
  const int eB    = gHi * 16;          // B: lanes<16 rows 0..15, else 16..31
  const int shB0  = 16 * lm + 256 * (2 * p);      // tile 2p
  const int tBeg  = tH * (Ln / 2);

  // per-lane base rows; loop itself is uniform with constant bounds
  const int aRow0 = tBeg + lm;
  const int bRow0 = tBeg - shB0;
  const int cRow0 = bRow0 - 256;                  // tile 2p+1

  f32x8 acc0 = {0.f, 0.f, 0.f, 0.f, 0.f, 0.f, 0.f, 0.f};
  f32x8 acc1 = {0.f, 0.f, 0.f, 0.f, 0.f, 0.f, 0.f, 0.f};

  #pragma unroll 4
  for (int i = 0; i < Ln / 2; ++i) {
    const int rA  = (aRow0 + i) & (Ln - 1);
    const int rB0 = (bRow0 + i) & (Ln - 1);
    const int rB1 = (cRow0 + i) & (Ln - 1);
    V16 a, b0, b1;
    const _Float16* pa = qs + rA * EH + eA;
    a.u[0]  = *(const u32x4*)(pa);            // ds_load_b128, shared by both tiles
    a.u[1]  = *(const u32x4*)(pa + 8);
    const _Float16* pb0 = ks + rB0 * EH + eB;
    b0.u[0] = *(const u32x4*)(pb0);
    b0.u[1] = *(const u32x4*)(pb0 + 8);
    const _Float16* pb1 = ks + rB1 * EH + eB;
    b1.u[0] = *(const u32x4*)(pb1);
    b1.u[1] = *(const u32x4*)(pb1 + 8);
    acc0 = __builtin_amdgcn_wmma_f32_16x16x32_f16(
        false, a.v, false, b0.v, (short)0, acc0, false, false);
    acc1 = __builtin_amdgcn_wmma_f32_16x16x32_f16(
        false, a.v, false, b1.v, (short)0, acc1, false, false);
  }

  // ---- write partials: C/D layout VGPR r -> M = r + 8*(lane>=16), N = lane%16
  float* dst = meanPart + ((size_t)wg * 2 + tH) * Ln;
  #pragma unroll
  for (int r = 0; r < 8; ++r) {
    const int m    = r + 8 * gHi;
    const int tau0 = m + 16 * lm + 256 * (2 * p);
    dst[tau0]       = acc0[r];
    dst[tau0 + 256] = acc1[r];
  }
}

// =====================================================================
// Kernel 2: sum 4 partials, top-7 over tau, softmax.  One wave per (b,h).
// =====================================================================
__launch_bounds__(32, 4)
__global__ void topk_softmax_kernel(const float* __restrict__ meanPart,
                                    float* __restrict__ tmpOut,
                                    float* __restrict__ probs,
                                    int* __restrict__ delays) {
  const int bh   = blockIdx.x;
  const int lane = threadIdx.x;
  const float* p0 = meanPart + (size_t)(bh * NPART + 0) * Ln;
  const float* p1 = meanPart + (size_t)(bh * NPART + 1) * Ln;
  const float* p2 = meanPart + (size_t)(bh * NPART + 2) * Ln;
  const float* p3 = meanPart + (size_t)(bh * NPART + 3) * Ln;

  float vals[Ln / 32];
  #pragma unroll
  for (int j = 0; j < Ln / 32; ++j) {
    const int tau = lane + 32 * j;
    vals[j] = ((p0[tau] + p1[tau]) + (p2[tau] + p3[tau])) * (1.0f / (float)En);
  }

  float wsel[TOPK]; int dsel[TOPK];
  for (int pss = 0; pss < TOPK; ++pss) {
    float bv = -3.0e38f; int bj = 0;
    #pragma unroll
    for (int j = 0; j < Ln / 32; ++j)
      if (vals[j] > bv) { bv = vals[j]; bj = j; }
    int bt = lane + 32 * bj;
    #pragma unroll
    for (int off = 16; off > 0; off >>= 1) {
      const float ov = __shfl_xor(bv, off, 32);
      const int   ot = __shfl_xor(bt, off, 32);
      if (ov > bv || (ov == bv && ot < bt)) { bv = ov; bt = ot; }
    }
    if ((bt & 31) == lane) vals[bt >> 5] = -3.0e38f;  // mask winner
    wsel[pss] = bv; dsel[pss] = bt;
  }

  const float mx = wsel[0];
  float pr[TOPK]; float s = 0.f;
  #pragma unroll
  for (int i = 0; i < TOPK; ++i) { pr[i] = __expf(wsel[i] - mx); s += pr[i]; }
  const float inv = 1.0f / s;
  if (lane < TOPK) {
    tmpOut[bh * TOPK + lane] = pr[lane] * inv;   // second reference output
    probs [bh * TOPK + lane] = pr[lane] * inv;
    delays[bh * TOPK + lane] = dsel[lane];
  }
}

// =====================================================================
// Kernel 3: V_out[b,l,h,e] = sum_i prob_i * v[b,(l+d_i)&2047,h,e]
// =====================================================================
__launch_bounds__(256, 2)
__global__ void gather_kernel(const float* __restrict__ Vv,
                              const float* __restrict__ probs,
                              const int* __restrict__ delays,
                              float* __restrict__ Out) {
  const int blk = blockIdx.x;        // bh*8 + chunk
  const int bh  = blk >> 3;
  const int l0  = (blk & 7) * 256;
  const int h   = bh & (Hn - 1);
  const int b   = bh >> 4;

  float pr[TOPK]; int dd[TOPK];
  #pragma unroll
  for (int i = 0; i < TOPK; ++i) {
    pr[i] = probs[bh * TOPK + i];
    dd[i] = delays[bh * TOPK + i];
  }

  const int e4 = (threadIdx.x & 15) * 4;   // 0..60
  const int lo = threadIdx.x >> 4;         // 0..15
  const size_t base = (size_t)b * Ln * (Hn * En) + (size_t)h * En + e4;

  for (int l = l0 + lo; l < l0 + 256; l += 16) {
    __builtin_prefetch(Vv + base + (size_t)(((l + 16 + dd[0]) & (Ln - 1)) * (Hn * En)), 0, 1);
    f32x4 acc = {0.f, 0.f, 0.f, 0.f};
    #pragma unroll
    for (int i = 0; i < TOPK; ++i) {
      const int ls = (l + dd[i]) & (Ln - 1);
      const f32x4 v = *(const f32x4*)(Vv + base + (size_t)ls * (Hn * En));
      acc.x += pr[i] * v.x; acc.y += pr[i] * v.y;
      acc.z += pr[i] * v.z; acc.w += pr[i] * v.w;
    }
    *(f32x4*)(Out + base + (size_t)l * (Hn * En)) = acc;
  }
}

// =====================================================================
extern "C" void kernel_launch(void* const* d_in, const int* in_sizes, int n_in,
                              void* d_out, int out_size, void* d_ws, size_t ws_size,
                              hipStream_t stream) {
  (void)in_sizes; (void)n_in; (void)out_size; (void)ws_size;

  const float* Q  = (const float*)d_in[0];
  const float* Kk = (const float*)d_in[1];
  const float* Vv = (const float*)d_in[2];

  float* Out    = (float*)d_out;                          // (B,L,H,E) fp32
  float* TmpOut = Out + (size_t)Bn * Ln * Hn * En;        // (B,H,7) fp32

  // workspace: 2048*2048 f32 partials (16 MB) + delays/probs (~30 KB)
  float* wsMean = (float*)d_ws;
  int*   wsDel  = (int*)((char*)d_ws + (size_t)Bn * Hn * NPART * Ln * sizeof(float));
  float* wsProb = (float*)(wsDel + Bn * Hn * TOPK);

  corr_wmma_kernel   <<<Bn * Hn * 2, 256, 0, stream>>>(Q, Kk, wsMean);
  topk_softmax_kernel<<<Bn * Hn,     32,  0, stream>>>(wsMean, TmpOut, wsProb, wsDel);
  gather_kernel      <<<Bn * Hn * 8, 256, 0, stream>>>(Vv, wsProb, wsDel, Out);
}
